// BaselineModel_56014963474646
// MI455X (gfx1250) — compile-verified, matrix operand order checked by reading, not run on hardware
//
#include <hip/hip_runtime.h>
#include <hip/hip_bf16.h>

// ---------------------------------------------------------------------------
// Spatio-temporal GAT + LSTM for MI455X (gfx1250, wave32).
// All GEMM operands pre-converted to bf16 ([N,K] weight layout so K is always
// contiguous); GEMM inner loop = global_load_b128 fragments + back-to-back
// v_wmma_f32_16x16x32_bf16 (f32 accumulate), 16x64 strip per wave.
// Irregular edge softmax/aggregation -> f32 global atomics (L2 resident).
// ---------------------------------------------------------------------------

#define DEV __device__ __forceinline__

typedef __attribute__((ext_vector_type(16))) __bf16 v16bf;
typedef __attribute__((ext_vector_type(8)))  __bf16 bf16x8;
typedef __attribute__((ext_vector_type(8)))  float  v8f;

static constexpr int Nn    = 500;    // nodes
static constexpr int Fin   = 32;     // input features
static constexpr int Hd    = 128;    // hidden dim
static constexpr int HEADS = 2;
static constexpr int Bb    = 4;
static constexpr int Tt    = 16;
static constexpr int Gg    = Bb * Tt;      // 64 graphs
static constexpr int Ee    = 8000;         // edges (w/o self loops)
static constexpr int Et    = Ee + Nn;      // 8500 with self loops
static constexpr int Mrows = Gg * Nn;      // 32000 (divisible by 64)
static constexpr int Seq   = Bb * Nn;      // 2000 LSTM sequences
static constexpr int SeqP  = 2048;         // Seq padded to a multiple of 64

// f32 -> bf16, round-to-nearest-even (bit manipulation; off the hot path).
DEV __bf16 f2bf(float f) {
  unsigned u = __builtin_bit_cast(unsigned, f);
  u += 0x7FFFu + ((u >> 16) & 1u);
  unsigned short h = (unsigned short)(u >> 16);
  return __builtin_bit_cast(__bf16, h);
}

// bf16 WMMA fragment when the 32 K-values are contiguous in memory.
// Per the ISA 16-bit 16x32 layout, lane (half = lane/16) needs
// k = e + 8*half (e<8) and k = 16 + (e-8) + 8*half (e>=8):
// two contiguous 8xbf16 runs -> two 128-bit loads.
DEV v16bf frag_bf(const __bf16* __restrict__ p, int half) {
  const bf16x8 lo = *(const bf16x8*)(p + 8 * half);
  const bf16x8 hi = *(const bf16x8*)(p + 16 + 8 * half);
  return __builtin_shufflevector(lo, hi, 0, 1, 2, 3, 4, 5, 6, 7,
                                 8, 9, 10, 11, 12, 13, 14, 15);
}

DEV float sigmoidf(float x) { return 1.f / (1.f + expf(-x)); }

DEV float atomicMaxF(float* addr, float val) {
  int* ia = (int*)addr;
  int old = __float_as_int(*addr);
  while (__int_as_float(old) < val) {
    int assumed = old;
    old = atomicCAS(ia, assumed, __float_as_int(val));
    if (old == assumed) break;
  }
  return __int_as_float(old);
}

// ---------------------------------------------------------------------------
// WMMA GEMM: C[M,N] = A[M,K] * B^T (+bias)(+relu)
//   A: bf16 [M,K] row-major; B: bf16 [N,K] row-major (K contiguous for both).
// Block = 128 threads = 4 waves along M; each wave computes a 16x64 strip
// (4 N-tiles, one shared A fragment, 4 v_wmma per K-step, K fully unrolled).
// Grid = (N/64, M/64); M % 64 == 0, N % 64 == 0; K is a template constant.
// ---------------------------------------------------------------------------
template <int K>
__global__ void gemm_wmma(const __bf16* __restrict__ A, const __bf16* __restrict__ B,
                          float* __restrict__ C, const float* __restrict__ bias,
                          int N, int relu)
{
  const int lane = threadIdx.x & 31;
  const int wave = threadIdx.x >> 5;
  const int tm   = blockIdx.y * 4 + wave;   // 16-row M tile
  const int n0   = blockIdx.x * 64;         // 64-col N strip
  const int half = lane >> 4;
  const int l16  = lane & 15;

  const __bf16* ap  = A + (size_t)(tm * 16 + l16) * K;
  const __bf16* bp0 = B + (size_t)(n0 + l16) * K;
  const __bf16* bp1 = B + (size_t)(n0 + 16 + l16) * K;
  const __bf16* bp2 = B + (size_t)(n0 + 32 + l16) * K;
  const __bf16* bp3 = B + (size_t)(n0 + 48 + l16) * K;

  v8f acc0 = {}, acc1 = {}, acc2 = {}, acc3 = {};
#pragma unroll
  for (int k0 = 0; k0 < K; k0 += 32) {
    const v16bf a  = frag_bf(ap + k0, half);
    const v16bf b0 = frag_bf(bp0 + k0, half);
    const v16bf b1 = frag_bf(bp1 + k0, half);
    const v16bf b2 = frag_bf(bp2 + k0, half);
    const v16bf b3 = frag_bf(bp3 + k0, half);
    acc0 = __builtin_amdgcn_wmma_f32_16x16x32_bf16(false, a, false, b0, (short)0, acc0, false, false);
    acc1 = __builtin_amdgcn_wmma_f32_16x16x32_bf16(false, a, false, b1, (short)0, acc1, false, false);
    acc2 = __builtin_amdgcn_wmma_f32_16x16x32_bf16(false, a, false, b2, (short)0, acc2, false, false);
    acc3 = __builtin_amdgcn_wmma_f32_16x16x32_bf16(false, a, false, b3, (short)0, acc3, false, false);
  }

#pragma unroll
  for (int u = 0; u < 4; ++u) {
    const v8f acc = (u == 0) ? acc0 : (u == 1) ? acc1 : (u == 2) ? acc2 : acc3;
    const int ocol = n0 + u * 16 + l16;
    const float badd = bias ? bias[ocol] : 0.f;
#pragma unroll
    for (int r = 0; r < 8; ++r) {
      const int orow = tm * 16 + 8 * half + r;  // C/D layout: VGPR r -> M=r(+8)
      float v = acc[r] + badd;
      if (relu) v = fmaxf(v, 0.f);
      C[(size_t)orow * N + ocol] = v;
    }
  }
}

// ---------------------------------------------------------------------------
// Operand conversion kernels (tiny, once per launch)
// ---------------------------------------------------------------------------
// in [Kd,Nd] f32 -> out [Nd,Kd] bf16 (transpose so K becomes contiguous)
__global__ void cvt_trans(const float* __restrict__ in, __bf16* __restrict__ out,
                          int Kd, int Nd)
{
  const int idx = blockIdx.x * blockDim.x + threadIdx.x;
  if (idx >= Kd * Nd) return;
  const int n = idx / Kd, k = idx % Kd;
  out[idx] = f2bf(in[(size_t)k * Nd + n]);
}

__global__ void cvt(const float* __restrict__ in, __bf16* __restrict__ out, int n)
{
  const int i = blockIdx.x * blockDim.x + threadIdx.x;
  if (i < n) out[i] = f2bf(in[i]);
}

// ---------------------------------------------------------------------------
// GATv2 edge phase (one wave per (graph, edge, head))
// ---------------------------------------------------------------------------
__global__ void gat_logits(const float* __restrict__ xl, const float* __restrict__ xr,
                           const int* __restrict__ ei, const float* __restrict__ att,
                           float* __restrict__ elog, float* __restrict__ nmax,
                           int heads, int Ctot)
{
  const int gw   = blockIdx.x * (blockDim.x >> 5) + (threadIdx.x >> 5);
  const int lane = threadIdx.x & 31;
  const int total = Gg * Et * heads;
  if (gw >= total) return;
  const int hd = gw % heads;
  const int e  = (gw / heads) % Et;
  const int g  = gw / (heads * Et);
  const int src = (e < Ee) ? ei[e] : (e - Ee);
  const int dst = (e < Ee) ? ei[Ee + e] : (e - Ee);
  const float* pl = xl + (size_t)(g * Nn + src) * Ctot + hd * Hd;
  const float* pr = xr + (size_t)(g * Nn + dst) * Ctot + hd * Hd;
  const float* pa = att + hd * Hd;
  float s = 0.f;
#pragma unroll
  for (int j = 0; j < Hd / 32; ++j) {
    const int c = lane + 32 * j;
    float v = pl[c] + pr[c];
    v = v > 0.f ? v : 0.2f * v;          // leaky_relu(0.2)
    s += v * pa[c];
  }
  for (int off = 16; off > 0; off >>= 1) s += __shfl_down(s, off, 32);
  if (lane == 0) {
    elog[gw] = s;
    atomicMaxF(&nmax[(g * Nn + dst) * heads + hd], s);
  }
}

__global__ void gat_expsum(float* __restrict__ elog, const int* __restrict__ ei,
                           const float* __restrict__ nmax, float* __restrict__ nsum,
                           int heads)
{
  const int idx = blockIdx.x * blockDim.x + threadIdx.x;
  const int total = Gg * Et * heads;
  if (idx >= total) return;
  const int hd = idx % heads;
  const int e  = (idx / heads) % Et;
  const int g  = idx / (heads * Et);
  const int dst = (e < Ee) ? ei[Ee + e] : (e - Ee);
  const float ex = expf(elog[idx] - nmax[(g * Nn + dst) * heads + hd]);
  elog[idx] = ex;
  atomicAdd(&nsum[(g * Nn + dst) * heads + hd], ex);
}

__global__ void gat_scatter(const float* __restrict__ xl, const int* __restrict__ ei,
                            const float* __restrict__ elog, const float* __restrict__ nsum,
                            float* __restrict__ agg, int heads, int Ctot)
{
  const int gw   = blockIdx.x * (blockDim.x >> 5) + (threadIdx.x >> 5);
  const int lane = threadIdx.x & 31;
  const int total = Gg * Et * heads;
  if (gw >= total) return;
  const int hd = gw % heads;
  const int e  = (gw / heads) % Et;
  const int g  = gw / (heads * Et);
  const int src = (e < Ee) ? ei[e] : (e - Ee);
  const int dst = (e < Ee) ? ei[Ee + e] : (e - Ee);
  const float alpha = elog[gw] / (nsum[(g * Nn + dst) * heads + hd] + 1e-16f);
  const float* pl = xl + (size_t)(g * Nn + src) * Ctot + hd * Hd;
  float* po = agg + (size_t)(g * Nn + dst) * Ctot + hd * Hd;
#pragma unroll
  for (int j = 0; j < Hd / 32; ++j) {
    const int c = lane + 32 * j;
    atomicAdd(&po[c], alpha * pl[c]);
  }
}

// ---------------------------------------------------------------------------
// BatchNorm(train stats over node axis) + ELU (+bias-in)(+residual).
// Reads f32 aggregation, writes bf16 (only consumer is a WMMA GEMM, so this
// is bit-identical to converting inside the GEMM). One block per (g, ch).
// ---------------------------------------------------------------------------
__global__ void bn_elu(const float* __restrict__ hin, __bf16* __restrict__ hout,
                       const float* __restrict__ gamma, const float* __restrict__ beta,
                       const float* __restrict__ preBias, const float* __restrict__ resid,
                       int C)
{
  const int g  = blockIdx.x / C;
  const int ch = blockIdx.x % C;
  const float* base = hin + (size_t)g * Nn * C + ch;
  const float pb = preBias ? preBias[ch] : 0.f;
  float s = 0.f, ss = 0.f;
  for (int i = threadIdx.x; i < Nn; i += blockDim.x) {
    const float v = base[(size_t)i * C] + pb;
    s += v; ss += v * v;
  }
  __shared__ float rs[256], rss[256];
  rs[threadIdx.x] = s; rss[threadIdx.x] = ss;
  __syncthreads();
  for (int off = blockDim.x >> 1; off > 0; off >>= 1) {
    if ((int)threadIdx.x < off) { rs[threadIdx.x] += rs[threadIdx.x + off];
                                  rss[threadIdx.x] += rss[threadIdx.x + off]; }
    __syncthreads();
  }
  const float mean = rs[0] / (float)Nn;
  const float var  = rss[0] / (float)Nn - mean * mean;   // biased, like jnp.var
  const float inv  = rsqrtf(var + 1e-5f) * gamma[ch];
  const float bb   = beta[ch];
  for (int i = threadIdx.x; i < Nn; i += blockDim.x) {
    const size_t off2 = (size_t)i * C;
    float v = (base[off2] + pb - mean) * inv + bb;
    v = v > 0.f ? v : expm1f(v);                          // ELU
    if (resid) v += resid[(size_t)g * Nn * C + off2];
    hout[(size_t)g * Nn * C + off2] = f2bf(v);
  }
}

// ---------------------------------------------------------------------------
// LSTM cell (gate order i,f,g,o). x-gate row = r*rs_r + t*rs_t.
// h written as bf16 (consumers: recurrent GEMM / FC1); c stays f32.
// ---------------------------------------------------------------------------
__global__ void lstm_cell(const float* __restrict__ xg, const float* __restrict__ rg,
                          const float* __restrict__ bih, const float* __restrict__ bhh,
                          __bf16* __restrict__ hst, float* __restrict__ cst,
                          __bf16* __restrict__ yout, int t, int rs_r, int rs_t)
{
  const int idx = blockIdx.x * blockDim.x + threadIdx.x;
  if (idx >= Seq * Hd) return;
  const int r = idx >> 7, j = idx & 127;
  const size_t xrow = (size_t)(r * rs_r + t * rs_t) * (4 * Hd);
  const size_t rrow = (size_t)r * (4 * Hd);
  const float gi = xg[xrow + j]          + rg[rrow + j]          + bih[j]          + bhh[j];
  const float gf = xg[xrow + Hd + j]     + rg[rrow + Hd + j]     + bih[Hd + j]     + bhh[Hd + j];
  const float gg = xg[xrow + 2 * Hd + j] + rg[rrow + 2 * Hd + j] + bih[2 * Hd + j] + bhh[2 * Hd + j];
  const float go = xg[xrow + 3 * Hd + j] + rg[rrow + 3 * Hd + j] + bih[3 * Hd + j] + bhh[3 * Hd + j];
  const float c  = sigmoidf(gf) * cst[idx] + sigmoidf(gi) * tanhf(gg);
  cst[idx] = c;
  const __bf16 hh = f2bf(sigmoidf(go) * tanhf(c));
  hst[idx] = hh;
  if (yout) yout[(size_t)t * Seq * Hd + idx] = hh;
}

// FC2: [2000,64] @ [64,1] + b; one wave per row.
__global__ void fc2_kernel(const float* __restrict__ a, const float* __restrict__ w,
                           const float* __restrict__ b2, float* __restrict__ out)
{
  const int gw   = blockIdx.x * (blockDim.x >> 5) + (threadIdx.x >> 5);
  const int lane = threadIdx.x & 31;
  if (gw >= Seq) return;
  const float* row = a + (size_t)gw * 64;
  float s = row[lane] * w[lane] + row[lane + 32] * w[lane + 32];
  for (int off = 16; off > 0; off >>= 1) s += __shfl_down(s, off, 32);
  if (lane == 0) out[gw] = s + b2[0];
}

__global__ void fill_kernel(float* __restrict__ p, float v, size_t n)
{
  const size_t i = (size_t)blockIdx.x * blockDim.x + threadIdx.x;
  if (i < n) p[i] = v;
}

// ---------------------------------------------------------------------------
extern "C" void kernel_launch(void* const* d_in, const int* in_sizes, int n_in,
                              void* d_out, int out_size, void* d_ws, size_t ws_size,
                              hipStream_t stream)
{
  (void)in_sizes; (void)n_in; (void)out_size; (void)ws_size;
  const float* x        = (const float*)d_in[0];
  const int*   ei       = (const int*)  d_in[1];
  const float* W_proj   = (const float*)d_in[2];
  const float* b_proj   = (const float*)d_in[3];
  const float* gat1_Wl  = (const float*)d_in[4];
  const float* gat1_Wr  = (const float*)d_in[5];
  const float* gat1_att = (const float*)d_in[6];
  const float* gat1_b   = (const float*)d_in[7];
  const float* bn1_g    = (const float*)d_in[8];
  const float* bn1_b    = (const float*)d_in[9];
  const float* gat2_Wl  = (const float*)d_in[10];
  const float* gat2_Wr  = (const float*)d_in[11];
  const float* gat2_att = (const float*)d_in[12];
  const float* gat2_b   = (const float*)d_in[13];
  const float* bn2_g    = (const float*)d_in[14];
  const float* bn2_b    = (const float*)d_in[15];
  const float* Wih0     = (const float*)d_in[16];
  const float* Whh0     = (const float*)d_in[17];
  const float* bih0     = (const float*)d_in[18];
  const float* bhh0     = (const float*)d_in[19];
  const float* Wih1     = (const float*)d_in[20];
  const float* Whh1     = (const float*)d_in[21];
  const float* bih1     = (const float*)d_in[22];
  const float* bhh1     = (const float*)d_in[23];
  const float* fc_W1    = (const float*)d_in[24];
  const float* fc_b1    = (const float*)d_in[25];
  const float* fc_W2    = (const float*)d_in[26];
  const float* fc_b2    = (const float*)d_in[27];
  float* out = (float*)d_out;

  // Workspace carve-up (byte allocator; everything kept 16B aligned)
  char* base = (char*)d_ws;
  size_t off = 0;
  auto allocf = [&](size_t n) { float* p = (float*)(base + off);
                                off = (off + n * 4 + 15) & ~(size_t)15; return p; };
  auto allocb = [&](size_t n) { __bf16* p = (__bf16*)(base + off);
                                off = (off + n * 2 + 15) & ~(size_t)15; return p; };
  // f32 buffers
  float* xl1  = allocf((size_t)Mrows * 2 * Hd);   // [32000,256]
  float* xr1  = allocf((size_t)Mrows * 2 * Hd);
  float* agg1 = allocf((size_t)Mrows * 2 * Hd);
  float* xl2  = allocf((size_t)Mrows * Hd);       // [32000,128]
  float* xr2  = allocf((size_t)Mrows * Hd);
  float* agg2 = allocf((size_t)Mrows * Hd);
  float* proj = allocf((size_t)Mrows * Hd);
  float* elog = allocf((size_t)Gg * Et * HEADS);
  float* nmax = allocf((size_t)Gg * Nn * HEADS);
  float* nsum = allocf((size_t)Gg * Nn * HEADS);
  float* xg   = allocf((size_t)Mrows * 4 * Hd);   // x-gates, reused across layers
  float* recg = allocf((size_t)SeqP * 4 * Hd);    // per-step recurrent gates
  float* cst  = allocf((size_t)SeqP * Hd);
  float* fc1o = allocf((size_t)SeqP * 64);
  // bf16 buffers (GEMM operands)
  __bf16* xbf   = allocb((size_t)Mrows * Fin);    // x
  __bf16* h1bf  = allocb((size_t)Mrows * 2 * Hd); // after BN1+ELU
  __bf16* hbf   = allocb((size_t)Mrows * Hd);     // Hbuf after BN2+ELU+resid
  __bf16* ys0bf = allocb((size_t)Tt * Seq * Hd);  // layer-0 outputs, time-major
  __bf16* hstbf = allocb((size_t)SeqP * Hd);      // padded; rows >= 2000 stay 0
  __bf16* w1l   = allocb((size_t)2 * Hd * Fin);   // [256,32]
  __bf16* w1r   = allocb((size_t)2 * Hd * Fin);
  __bf16* w2l   = allocb((size_t)Hd * 2 * Hd);    // [128,256]
  __bf16* w2r   = allocb((size_t)Hd * 2 * Hd);
  __bf16* wpj   = allocb((size_t)Hd * Fin);       // [128,32]
  __bf16* wf1   = allocb((size_t)64 * Hd);        // [64,128]
  __bf16* wi0   = allocb((size_t)4 * Hd * Hd);    // [512,128] (already [N,K])
  __bf16* wh0   = allocb((size_t)4 * Hd * Hd);
  __bf16* wi1   = allocb((size_t)4 * Hd * Hd);
  __bf16* wh1   = allocb((size_t)4 * Hd * Hd);

  const dim3 B128(128), B256(256);
  auto fills = [&](float* p, float v, size_t n) {
    fill_kernel<<<(unsigned)((n + 255) / 256), B256, 0, stream>>>(p, v, n);
  };
  auto cvtT = [&](const float* in, __bf16* o2, int Kd, int Nd) {
    cvt_trans<<<(Kd * Nd + 255) / 256, B256, 0, stream>>>(in, o2, Kd, Nd);
  };
  auto cvtS = [&](const float* in, __bf16* o2, int n) {
    cvt<<<(n + 255) / 256, B256, 0, stream>>>(in, o2, n);
  };

  // ---- operand conversion (tiny) ------------------------------------------
  cvtS(x, xbf, Mrows * Fin);
  cvtT(gat1_Wl, w1l, Fin, 2 * Hd);
  cvtT(gat1_Wr, w1r, Fin, 2 * Hd);
  cvtT(gat2_Wl, w2l, 2 * Hd, Hd);
  cvtT(gat2_Wr, w2r, 2 * Hd, Hd);
  cvtT(W_proj, wpj, Fin, Hd);
  cvtT(fc_W1, wf1, Hd, 64);
  cvtS(Wih0, wi0, 4 * Hd * Hd);
  cvtS(Whh0, wh0, 4 * Hd * Hd);
  cvtS(Wih1, wi1, 4 * Hd * Hd);
  cvtS(Whh1, wh1, 4 * Hd * Hd);

  // ---------------- GAT layer 1 (F=32 -> 2 heads x 128, concat) ------------
  fills(agg1, 0.f, (size_t)Mrows * 2 * Hd);
  fills(nmax, -3.0e38f, (size_t)Gg * Nn * HEADS);
  fills(nsum, 0.f, (size_t)Gg * Nn * HEADS);
  gemm_wmma<Fin><<<dim3((2 * Hd) / 64, Mrows / 64), B128, 0, stream>>>(xbf, w1l, xl1, nullptr, 2 * Hd, 0);
  gemm_wmma<Fin><<<dim3((2 * Hd) / 64, Mrows / 64), B128, 0, stream>>>(xbf, w1r, xr1, nullptr, 2 * Hd, 0);
  {
    const int tot = Gg * Et * HEADS;
    gat_logits<<<(tot + 7) / 8, B256, 0, stream>>>(xl1, xr1, ei, gat1_att, elog, nmax, HEADS, 2 * Hd);
    gat_expsum<<<(tot + 255) / 256, B256, 0, stream>>>(elog, ei, nmax, nsum, HEADS);
    gat_scatter<<<(tot + 7) / 8, B256, 0, stream>>>(xl1, ei, elog, nsum, agg1, HEADS, 2 * Hd);
  }
  bn_elu<<<Gg * 2 * Hd, B256, 0, stream>>>(agg1, h1bf, bn1_g, bn1_b, gat1_b, nullptr, 2 * Hd);

  // ---------------- GAT layer 2 (256 -> 1 head x 128) ----------------------
  gemm_wmma<2 * Hd><<<dim3(Hd / 64, Mrows / 64), B128, 0, stream>>>(h1bf, w2l, xl2, nullptr, Hd, 0);
  gemm_wmma<2 * Hd><<<dim3(Hd / 64, Mrows / 64), B128, 0, stream>>>(h1bf, w2r, xr2, nullptr, Hd, 0);
  fills(agg2, 0.f, (size_t)Mrows * Hd);
  fills(nmax, -3.0e38f, (size_t)Gg * Nn);
  fills(nsum, 0.f, (size_t)Gg * Nn);
  {
    const int tot = Gg * Et;
    gat_logits<<<(tot + 7) / 8, B256, 0, stream>>>(xl2, xr2, ei, gat2_att, elog, nmax, 1, Hd);
    gat_expsum<<<(tot + 255) / 256, B256, 0, stream>>>(elog, ei, nmax, nsum, 1);
    gat_scatter<<<(tot + 7) / 8, B256, 0, stream>>>(xl2, ei, elog, nsum, agg2, 1, Hd);
  }
  gemm_wmma<Fin><<<dim3(Hd / 64, Mrows / 64), B128, 0, stream>>>(xbf, wpj, proj, b_proj, Hd, 0);
  bn_elu<<<Gg * Hd, B256, 0, stream>>>(agg2, hbf, bn2_g, bn2_b, gat2_b, proj, Hd);
  // hbf = Hbuf [32000,128] bf16, flat in (b,t,n) order.

  // ---------------- LSTM layer 0 -------------------------------------------
  // x-gates for all (r,t): row m = r*16 + t of Hbuf (raw reshape semantics).
  gemm_wmma<Hd><<<dim3((4 * Hd) / 64, Mrows / 64), B128, 0, stream>>>(hbf, wi0, xg, nullptr, 4 * Hd, 0);
  fills((float*)hstbf, 0.f, (size_t)SeqP * Hd / 2);  // bf16 zeros; pad rows stay 0
  fills(cst, 0.f, (size_t)SeqP * Hd);
  for (int t = 0; t < Tt; ++t) {
    gemm_wmma<Hd><<<dim3((4 * Hd) / 64, SeqP / 64), B128, 0, stream>>>(hstbf, wh0, recg, nullptr, 4 * Hd, 0);
    lstm_cell<<<(Seq * Hd + 255) / 256, B256, 0, stream>>>(
        xg, recg, bih0, bhh0, hstbf, cst, ys0bf, t, /*rs_r=*/Tt, /*rs_t=*/1);
  }

  // ---------------- LSTM layer 1 -------------------------------------------
  // ys0 is time-major [t,2000,128]: x-gate row m = t*2000 + r.
  gemm_wmma<Hd><<<dim3((4 * Hd) / 64, Mrows / 64), B128, 0, stream>>>(ys0bf, wi1, xg, nullptr, 4 * Hd, 0);
  fills((float*)hstbf, 0.f, (size_t)SeqP * Hd / 2);
  fills(cst, 0.f, (size_t)SeqP * Hd);
  for (int t = 0; t < Tt; ++t) {
    gemm_wmma<Hd><<<dim3((4 * Hd) / 64, SeqP / 64), B128, 0, stream>>>(hstbf, wh1, recg, nullptr, 4 * Hd, 0);
    lstm_cell<<<(Seq * Hd + 255) / 256, B256, 0, stream>>>(
        xg, recg, bih1, bhh1, hstbf, cst, nullptr, t, /*rs_r=*/1, /*rs_t=*/Seq);
  }

  // ---------------- Head ----------------------------------------------------
  gemm_wmma<Hd><<<dim3(64 / 64, SeqP / 64), B128, 0, stream>>>(hstbf, wf1, fc1o, fc_b1, 64, /*relu=*/1);
  fc2_kernel<<<(Seq + 7) / 8, B256, 0, stream>>>(fc1o, fc_W2, fc_b2, out);
}
